// MultiheadAttention_16939351015974
// MI455X (gfx1250) — matrix operand order repaired and checked by reading, not compiled, not run on hardware
//
#include <hip/hip_runtime.h>
#include <hip/hip_bf16.h>
#include <cstdint>

typedef __attribute__((ext_vector_type(16))) __bf16 v16bf;
typedef __attribute__((ext_vector_type(8)))  __bf16 v8bf;
typedef __attribute__((ext_vector_type(8)))  float  v8f;

#define WMMA_BF16(A,B,C) __builtin_amdgcn_wmma_f32_16x16x32_bf16(false,(A),false,(B),(short)0,(C),false,false)

constexpr int BSZ = 2;
constexpr int T   = 2048;
constexpr int E   = 512;
constexpr int H   = 16;
constexpr int E3  = 1536;
constexpr float SCALE = 0.17677669529663687f;   // 1/sqrt(32)

__device__ __forceinline__ v16bf cat16(v8bf lo, v8bf hi) {
  return __builtin_shufflevector(lo, hi, 0,1,2,3,4,5,6,7,8,9,10,11,12,13,14,15);
}
__device__ __forceinline__ float rmax16(float v) {
  v = fmaxf(v, __shfl_xor(v, 1));
  v = fmaxf(v, __shfl_xor(v, 2));
  v = fmaxf(v, __shfl_xor(v, 4));
  v = fmaxf(v, __shfl_xor(v, 8));
  return v;
}
__device__ __forceinline__ float rsum16(float v) {
  v += __shfl_xor(v, 1);
  v += __shfl_xor(v, 2);
  v += __shfl_xor(v, 4);
  v += __shfl_xor(v, 8);
  return v;
}
// gfx1250 async global->LDS copy (per-lane B128), tracked by ASYNCcnt.
__device__ __forceinline__ void async_load_b128(unsigned lds, const void* g) {
  asm volatile("global_load_async_to_lds_b128 %0, %1, off" :: "v"(lds), "v"(g) : "memory");
}

// ---------------- Kernel 1: qkv = x @ w_qkv^T + b_qkv (bf16 out) ----------------
// Wave computes a 64x64 tile (4x4 WMMA grid). M = 4096, N = 1536, K = 512.
__global__ __launch_bounds__(256) void qkv_gemm(const float* __restrict__ x,
                                                const float* __restrict__ w,
                                                const float* __restrict__ bias,
                                                __bf16* __restrict__ y) {
  const int lane = threadIdx.x & 31;
  const int ln   = lane & 15;
  const int kh   = lane >> 4;
  const int wid  = (blockIdx.x * 256 + threadIdx.x) >> 5;   // 0..1535
  const int m0   = (wid / 24) * 64;
  const int n0   = (wid % 24) * 64;

  v8f acc[4][4] = {};
  for (int kk = 0; kk < E; kk += 32) {
    v16bf a[4];
    #pragma unroll
    for (int ms = 0; ms < 4; ++ms) {
      const float* xp = x + (size_t)(m0 + ms * 16 + ln) * E + kk + 16 * kh;
      #pragma unroll
      for (int j = 0; j < 8; ++j)  a[ms][j]     = (__bf16)xp[j];
      #pragma unroll
      for (int j = 0; j < 8; ++j)  a[ms][8 + j] = (__bf16)xp[8 + j];
    }
    #pragma unroll
    for (int t = 0; t < 4; ++t) {
      const float* wp = w + (size_t)(n0 + t * 16 + ln) * E + kk + 16 * kh;
      v16bf bm;
      #pragma unroll
      for (int j = 0; j < 16; ++j) bm[j] = (__bf16)wp[j];
      #pragma unroll
      for (int ms = 0; ms < 4; ++ms)
        acc[ms][t] = WMMA_BF16(a[ms], bm, acc[ms][t]);
    }
  }
  #pragma unroll
  for (int t = 0; t < 4; ++t) {
    const int n = n0 + t * 16 + ln;
    const float bv = bias[n];
    #pragma unroll
    for (int ms = 0; ms < 4; ++ms) {
      #pragma unroll
      for (int r = 0; r < 8; ++r) {
        const int m = m0 + ms * 16 + r + 8 * kh;
        y[(size_t)m * E3 + n] = (__bf16)(acc[ms][t][r] + bv);
      }
    }
  }
}

// ---------------- Kernel 2: attention for one (b, 16-row tile), all heads ----------------
// 8 waves split S (256 cols each). Mask read exactly once (nontemporal).
// V chunks staged to LDS via async B128 DMA, double-buffered under compute.
__global__ __launch_bounds__(256, 1) void attn(const float* __restrict__ mask,
                                               const __bf16* __restrict__ qkv,
                                               __bf16* __restrict__ o_ws,
                                               float* __restrict__ aavg) {
  // LDS arena: [0,512) rowred | [1024,9216) P chunk strips (1K/wave)
  //            [16384,49152) V stage (2KB x 2 bufs / wave) | [49152,65536) O partials
  __shared__ __align__(16) unsigned char arena[65536];

  const int lane = threadIdx.x & 31;
  const int ln   = lane & 15;
  const int kh   = lane >> 4;
  const int wave = threadIdx.x >> 5;
  const int b    = blockIdx.x >> 7;
  const int t0   = (blockIdx.x & 127) << 4;
  const int s0   = wave * 256;

  float* rowred      = (float*)arena;                                   // [8][16]
  __bf16* pstrip     = (__bf16*)(arena + 1024) + wave * 512;            // [16][32]
  __bf16* vst        = (__bf16*)(arena + 16384) + wave * 2048;          // 2 x [32][32]
  float* opart       = (float*)(arena + 49152);                         // [8][16][32]
  const unsigned vst_lds = (unsigned)(uintptr_t)(arena + 16384 + wave * 4096);

  v8f av[16] = {};                            // a_avg accumulator across heads

  #pragma clang loop unroll(disable)
  for (int h = 0; h < H; ++h) {
    const __bf16* vbase = qkv + (size_t)(b * T) * E3 + 2 * E + h * 32;
    // async-stage a 32x32 V chunk into LDS buffer (cc&1): 4 per-lane B128 ops
    auto issue_chunk = [&](int cc) {
      #pragma unroll
      for (int i = 0; i < 4; ++i) {
        const int idx  = i * 32 + lane;     // 0..127
        const int row  = idx >> 2;          // s within chunk, 0..31
        const int part = idx & 3;           // 16B slice of the 64B row
        const __bf16* g = vbase + (size_t)(s0 + cc * 32 + row) * E3;
        const unsigned lds = vst_lds + ((cc & 1) << 11) + row * 64 + part * 16;
        async_load_b128(lds, (const void*)((const char*)g + part * 16));
      }
    };
    issue_chunk(0);                         // prefetch V under scores+softmax
    issue_chunk(1);

    // ---- Q fragment (16x32) ----
    const __bf16* qp = qkv + ((size_t)(b * T + t0 + ln)) * E3 + h * 32 + 8 * kh;
    v16bf qa = cat16(*(const v8bf*)qp, *(const v8bf*)(qp + 16));

    // ---- scores: S' = mask + SCALE * Q K^T ----
    v8f sp[16];
    const float* mrow = mask + ((size_t)(b * H + h) * T + t0) * T + s0;
    #pragma unroll
    for (int ts = 0; ts < 16; ++ts) {
      const int s = s0 + ts * 16 + ln;
      const __bf16* kp = qkv + (size_t)(b * T + s) * E3 + E + h * 32 + 8 * kh;
      v16bf kb = cat16(*(const v8bf*)kp, *(const v8bf*)(kp + 16));
      v8f c = {};
      c = WMMA_BF16(qa, kb, c);
      if (ts < 15) __builtin_prefetch(&mrow[(size_t)(8 * kh) * T + (ts + 1) * 16 + ln], 0, 0);
      v8f sv;
      #pragma unroll
      for (int r = 0; r < 8; ++r) {
        const float mv = __builtin_nontemporal_load(&mrow[(size_t)(r + 8 * kh) * T + ts * 16 + ln]);
        sv[r] = mv + SCALE * c[r];
      }
      sp[ts] = sv;
    }

    // ---- row max: shfl within 16-lane group, LDS across waves ----
    float m[8];
    #pragma unroll
    for (int r = 0; r < 8; ++r) {
      float v = sp[0][r];
      #pragma unroll
      for (int ts = 1; ts < 16; ++ts) v = fmaxf(v, sp[ts][r]);
      m[r] = rmax16(v);
    }
    if (ln == 0) {
      #pragma unroll
      for (int r = 0; r < 8; ++r) rowred[wave * 16 + r + 8 * kh] = m[r];
    }
    __syncthreads();
    #pragma unroll
    for (int r = 0; r < 8; ++r) {
      float v = rowred[r + 8 * kh];
      #pragma unroll
      for (int w = 1; w < 8; ++w) v = fmaxf(v, rowred[w * 16 + r + 8 * kh]);
      m[r] = v;
    }
    __syncthreads();

    // ---- exp + row sum ----
    float l[8];
    #pragma unroll
    for (int r = 0; r < 8; ++r) l[r] = 0.f;
    #pragma unroll
    for (int ts = 0; ts < 16; ++ts) {
      #pragma unroll
      for (int r = 0; r < 8; ++r) {
        const float e = __expf(sp[ts][r] - m[r]);
        sp[ts][r] = e;
        l[r] += e;
      }
    }
    #pragma unroll
    for (int r = 0; r < 8; ++r) l[r] = rsum16(l[r]);
    if (ln == 0) {
      #pragma unroll
      for (int r = 0; r < 8; ++r) rowred[wave * 16 + r + 8 * kh] = l[r];
    }
    __syncthreads();
    float inv[8];
    #pragma unroll
    for (int r = 0; r < 8; ++r) {
      float v = rowred[r + 8 * kh];
      #pragma unroll
      for (int w = 1; w < 8; ++w) v += rowred[w * 16 + r + 8 * kh];
      inv[r] = 1.0f / v;
    }
    __syncthreads();

    // ---- normalize P in regs, accumulate a_avg ----
    #pragma unroll
    for (int ts = 0; ts < 16; ++ts) {
      #pragma unroll
      for (int r = 0; r < 8; ++r) {
        const float p = sp[ts][r] * inv[r];
        av[ts][r] += p;
        sp[ts][r] = p;
      }
    }

    // ---- O partial = P(16x256) @ V(256x32): 8 K-chunks, async double buffer ----
    v8f oa0 = {}, oa1 = {};
    #pragma unroll
    for (int cc = 0; cc < 8; ++cc) {
      // P chunk -> LDS strip (C-layout -> A-layout transpose through LDS)
      #pragma unroll
      for (int tt = 0; tt < 2; ++tt) {
        const int ts = cc * 2 + tt;
        #pragma unroll
        for (int r = 0; r < 8; ++r)
          pstrip[(r + 8 * kh) * 32 + tt * 16 + ln] = (__bf16)sp[ts][r];
      }
      const __bf16* pr = pstrip + ln * 32 + 16 * kh;
      v16bf pa = cat16(*(const v8bf*)pr, *(const v8bf*)(pr + 8));

      // wait for this chunk's async V stage (next chunk may stay in flight)
      if (cc < 7) { asm volatile("s_wait_asynccnt 4" ::: "memory"); }
      else        { asm volatile("s_wait_asynccnt 0" ::: "memory"); }
      const __bf16* vb = vst + (cc & 1) * 1024;
      v16bf vb0, vb1;
      #pragma unroll
      for (int j = 0; j < 16; ++j) {
        vb0[j] = vb[(j + 16 * kh) * 32 + ln];
        vb1[j] = vb[(j + 16 * kh) * 32 + 16 + ln];
      }
      oa0 = WMMA_BF16(pa, vb0, oa0);
      oa1 = WMMA_BF16(pa, vb1, oa1);
      if (cc < 6) {
        // drain DS pipe before async engine may overwrite this buffer
        asm volatile("s_wait_dscnt 0" ::: "memory");
        issue_chunk(cc + 2);
      }
    }

    // ---- cross-wave O reduction, store bf16 o tile ----
    __syncthreads();
    #pragma unroll
    for (int r = 0; r < 8; ++r) {
      opart[(wave * 16 + r + 8 * kh) * 32 + ln]      = oa0[r];
      opart[(wave * 16 + r + 8 * kh) * 32 + 16 + ln] = oa1[r];
    }
    __syncthreads();
    for (int i = threadIdx.x; i < 512; i += 256) {
      const int M = i >> 5, d = i & 31;
      float s = 0.f;
      #pragma unroll
      for (int w = 0; w < 8; ++w) s += opart[(w * 16 + M) * 32 + d];
      o_ws[(size_t)(b * T + t0 + M) * E + h * 32 + d] = (__bf16)s;
    }
    __syncthreads();
  }

  // ---- store a_avg (sum over heads / 16), streaming stores ----
  #pragma unroll
  for (int ts = 0; ts < 16; ++ts) {
    #pragma unroll
    for (int r = 0; r < 8; ++r) {
      const int row = t0 + r + 8 * kh;
      const int col = s0 + ts * 16 + ln;
      __builtin_nontemporal_store(av[ts][r] * 0.0625f,
                                  &aavg[((size_t)(b * T) + row) * T + col]);
    }
  }
}

// ---------------- Kernel 3: out = o @ w_o^T + b_o (fp32 out), 64x64 tiles ----------------
__global__ __launch_bounds__(256) void out_gemm(const __bf16* __restrict__ o,
                                                const float* __restrict__ w,
                                                const float* __restrict__ bias,
                                                float* __restrict__ out) {
  const int lane = threadIdx.x & 31;
  const int ln   = lane & 15;
  const int kh   = lane >> 4;
  const int wid  = (blockIdx.x * 256 + threadIdx.x) >> 5;   // 0..511
  const int m0   = (wid >> 3) * 64;
  const int n0   = (wid & 7) * 64;

  v8f acc[4][4] = {};
  for (int kk = 0; kk < E; kk += 32) {
    v16bf a[4];
    #pragma unroll
    for (int ms = 0; ms < 4; ++ms) {
      const __bf16* opr = o + (size_t)(m0 + ms * 16 + ln) * E + kk + 16 * kh;
      a[ms] = cat16(*(const v8bf*)opr, *(const v8bf*)(opr + 8));
    }
    #pragma unroll
    for (int t = 0; t < 4; ++t) {
      const float* wp = w + (size_t)(n0 + t * 16 + ln) * E + kk + 16 * kh;
      v16bf bm;
      #pragma unroll
      for (int j = 0; j < 16; ++j) bm[j] = (__bf16)wp[j];
      #pragma unroll
      for (int ms = 0; ms < 4; ++ms)
        acc[ms][t] = WMMA_BF16(a[ms], bm, acc[ms][t]);
    }
  }
  #pragma unroll
  for (int t = 0; t < 4; ++t) {
    const int n = n0 + t * 16 + ln;
    const float bv = bias[n];
    #pragma unroll
    for (int ms = 0; ms < 4; ++ms) {
      #pragma unroll
      for (int r = 0; r < 8; ++r) {
        const int m = m0 + ms * 16 + r + 8 * kh;
        out[(size_t)m * E + n] = acc[ms][t][r] + bv;
      }
    }
  }
}

extern "C" void kernel_launch(void* const* d_in, const int* in_sizes, int n_in,
                              void* d_out, int out_size, void* d_ws, size_t ws_size,
                              hipStream_t stream) {
  const float* x    = (const float*)d_in[0];
  const float* mask = (const float*)d_in[1];
  const float* wqkv = (const float*)d_in[2];
  const float* bqkv = (const float*)d_in[3];
  const float* wo   = (const float*)d_in[4];
  const float* bo   = (const float*)d_in[5];

  float* out_o = (float*)d_out;                               // [B*T, E]
  float* out_a = out_o + (size_t)BSZ * T * E;                 // [B, T, T]

  __bf16* qkv_ws = (__bf16*)d_ws;                             // [B*T, 3E] bf16
  __bf16* o_ws   = qkv_ws + (size_t)BSZ * T * E3;             // [B*T, E]  bf16

  // 1) packed QKV projection: 64x64 tiles -> 1536 waves
  qkv_gemm<<<192, 256, 0, stream>>>(x, wqkv, bqkv, qkv_ws);
  // 2) attention: one workgroup per (b, 16 query rows)
  attn<<<BSZ * (T / 16), 256, 0, stream>>>(mask, qkv_ws, o_ws, out_a);
  // 3) output projection: 64x64 tiles -> 512 waves
  out_gemm<<<64, 256, 0, stream>>>(o_ws, wo, bo, out_o);
}
